// DilatedAttentionNew_25975962206459
// MI455X (gfx1250) — matrix operand order repaired and checked by reading, not compiled, not run on hardware
//
#include <hip/hip_runtime.h>

typedef _Float16 v16h __attribute__((ext_vector_type(16)));
typedef _Float16 h8   __attribute__((ext_vector_type(8)));
typedef _Float16 h4   __attribute__((ext_vector_type(4)));
typedef float    v8f  __attribute__((ext_vector_type(8)));

#define DDIM   768
#define NTOK   8192
#define SEGW   512
#define QT     32        // query rows per workgroup
#define DSTEP  32        // d-chunk for phase 1 (WMMA K)
#define KC     32        // key chunk for phase 2 (WMMA K)

// padded strides (elements)
#define QSTR   776       // f16 row stride for Q tile (16B-aligned rows)
#define KCSTR  264       // f16 row stride for K^T chunk rows (d-major)
#define SSTR   264       // f32 row stride for scores
#define PSTR   264       // f16 row stride for probabilities
#define VSTR   776       // f16 row stride for V chunk rows

// LDS layout (bytes); Q+Kcol (phase1) / Sbuf (softmax) / Vrow (phase2) alias at 0
#define QBUF_BYTES (QT * QSTR * 2)        // 49664
#define KCOL_OFF   QBUF_BYTES
#define KCOL_BYTES (DSTEP * KCSTR * 2)    // 16896
#define PBUF_OFF   (QBUF_BYTES + KCOL_BYTES)  // 66560
#define PBUF_BYTES (QT * PSTR * 2)        // 16896
#define SMEM_BYTES (PBUF_OFF + PBUF_BYTES)    // 83456

static __device__ __forceinline__ v16h ld_a16(const _Float16* p) {
    // A-operand per-lane data: two contiguous 16B chunks (k .. k+7, k+16 .. k+23)
    h8 lo = *(const h8*)p;
    h8 hi = *(const h8*)(p + 16);
    return __builtin_shufflevector(lo, hi, 0,1,2,3,4,5,6,7,8,9,10,11,12,13,14,15);
}
static __device__ __forceinline__ v16h ld_b16(const _Float16* p, int stride) {
    // B-operand per-lane data: one B column, k rows (k .. k+7, k+16 .. k+23) strided by stride
    v16h r;
    #pragma unroll
    for (int s = 0; s < 8; ++s) { r[s] = p[s * stride]; r[s + 8] = p[(16 + s) * stride]; }
    return r;
}

__global__ __launch_bounds__(256)
void dilated_attn_f16_wmma(const float* __restrict__ x, float* __restrict__ y)
{
    __shared__ __align__(16) unsigned char smem[SMEM_BYTES];
    _Float16* Qbuf = (_Float16*)(smem);              // phase 1
    _Float16* Kcol = (_Float16*)(smem + KCOL_OFF);   // phase 1 (K^T chunk, [d][key])
    float*    Sbuf = (float*)(smem);                 // softmax (aliases Qbuf/Kcol)
    _Float16* Vrow = (_Float16*)(smem);              // phase 2 (aliases)
    _Float16* Pbuf = (_Float16*)(smem + PBUF_OFF);   // probabilities

    const int t    = threadIdx.x;
    const int wave = t >> 5;
    const int lane = t & 31;
    const int half = lane >> 4;
    const int l16  = lane & 15;

    const int wg  = blockIdx.x;          // 4 * 16 * 8 = 512
    const int b   = wg >> 7;
    const int seg = (wg >> 3) & 15;
    const int qt  = wg & 7;
    const int q0  = qt * QT;

    const float scale = 0.03608439182435161f;  // 768^-0.5
    const size_t segbase = ((size_t)b * NTOK + (size_t)seg * SEGW) * DDIM;

    // ---------------- stage Q tile (f16, pre-scaled) ----------------
    {
        const int r  = t >> 3;           // query row 0..31
        const int cg = t & 7;            // column group
        const float4* src = (const float4*)(x + segbase + (size_t)(2 * (q0 + r)) * DDIM);
        _Float16* dst = Qbuf + r * QSTR;
        #pragma unroll
        for (int i = 0; i < 24; ++i) {
            int c4 = cg + i * 8;         // float4 index 0..191 (coalesced)
            float4 v = src[c4];
            h4 hv = { (_Float16)(v.x * scale), (_Float16)(v.y * scale),
                      (_Float16)(v.z * scale), (_Float16)(v.w * scale) };
            *(h4*)(dst + c4 * 4) = hv;
        }
    }

    // ---------------- phase 1: S = Q K^T (f32 accum in registers) ----------------
    v8f accS[2][2];
    #pragma unroll
    for (int m = 0; m < 2; ++m)
        #pragma unroll
        for (int n = 0; n < 2; ++n)
            accS[m][n] = (v8f){0.f,0.f,0.f,0.f,0.f,0.f,0.f,0.f};

    const int myn0 = wave * 2;           // this wave owns score n-tiles {2w, 2w+1}

    for (int ds = 0; ds < DDIM / DSTEP; ++ds) {
        const int d0 = ds * DSTEP;
        __syncthreads();                 // protect Kcol reuse (and Q staging on first iter)
        // stage K^T chunk: Kcol[dlocal][key] = x[row(2*key)][d0+dlocal]
        // lane = dlocal -> coalesced 128B global reads, low-conflict LDS stores
        #pragma unroll 8
        for (int it = 0; it < 32; ++it) {
            int key = wave + it * 8;
            float v = x[segbase + (size_t)(2 * key) * DDIM + d0 + lane];
            Kcol[lane * KCSTR + key] = (_Float16)v;
        }
        __syncthreads();

        v16h A[2];
        #pragma unroll
        for (int m = 0; m < 2; ++m)
            A[m] = ld_a16(Qbuf + (m * 16 + l16) * QSTR + d0 + half * 8);

        #pragma unroll
        for (int nt = 0; nt < 2; ++nt) {
            v16h B = ld_b16(Kcol + half * 8 * KCSTR + (myn0 + nt) * 16 + l16, KCSTR);
            accS[0][nt] = __builtin_amdgcn_wmma_f32_16x16x32_f16(
                false, A[0], false, B, (short)0, accS[0][nt], false, false);
            accS[1][nt] = __builtin_amdgcn_wmma_f32_16x16x32_f16(
                false, A[1], false, B, (short)0, accS[1][nt], false, false);
        }
    }

    // ---------------- spill S to LDS (aliases dead Q/Kcol) ----------------
    __syncthreads();
    #pragma unroll
    for (int m = 0; m < 2; ++m)
        #pragma unroll
        for (int nt = 0; nt < 2; ++nt)
            #pragma unroll
            for (int v = 0; v < 8; ++v) {
                int row = m * 16 + half * 8 + v;         // C layout: VGPR v -> M=v (+8 for hi half)
                int col = (myn0 + nt) * 16 + l16;
                Sbuf[row * SSTR + col] = accS[m][nt][v];
            }
    __syncthreads();

    // ---------------- softmax (8 lanes per row) ----------------
    {
        const int row = t >> 3;
        const int c0  = (t & 7) * 32;
        const float* sp = Sbuf + row * SSTR + c0;
        float mx = -1e30f;
        float e[32];
        #pragma unroll
        for (int i = 0; i < 32; ++i) mx = fmaxf(mx, sp[i]);
        mx = fmaxf(mx, __shfl_xor(mx, 1));
        mx = fmaxf(mx, __shfl_xor(mx, 2));
        mx = fmaxf(mx, __shfl_xor(mx, 4));
        float sum = 0.f;
        #pragma unroll
        for (int i = 0; i < 32; ++i) { e[i] = __expf(sp[i] - mx); sum += e[i]; }
        sum += __shfl_xor(sum, 1);
        sum += __shfl_xor(sum, 2);
        sum += __shfl_xor(sum, 4);
        float inv = 1.0f / sum;
        _Float16* pp = Pbuf + row * PSTR + c0;
        #pragma unroll
        for (int i = 0; i < 32; ++i) pp[i] = (_Float16)(e[i] * inv);
    }
    __syncthreads();

    // ---------------- phase 2: O = P V (f32 accum in registers) ----------------
    v8f accO[2][6];
    #pragma unroll
    for (int m = 0; m < 2; ++m)
        #pragma unroll
        for (int n = 0; n < 6; ++n)
            accO[m][n] = (v8f){0.f,0.f,0.f,0.f,0.f,0.f,0.f,0.f};

    for (int kc = 0; kc < 256 / KC; ++kc) {
        __syncthreads();                 // protect Vrow reuse (and Sbuf alias on first iter)
        // stage V chunk rows (f16, row-major [key][d])
        {
            const int r  = t >> 3;       // key row within chunk
            const int cg = t & 7;
            const float4* src = (const float4*)(x + segbase + (size_t)(2 * (kc * KC + r)) * DDIM);
            _Float16* dst = Vrow + r * VSTR;
            #pragma unroll
            for (int i = 0; i < 24; ++i) {
                int c4 = cg + i * 8;
                float4 v = src[c4];
                h4 hv = { (_Float16)v.x, (_Float16)v.y, (_Float16)v.z, (_Float16)v.w };
                *(h4*)(dst + c4 * 4) = hv;
            }
        }
        __syncthreads();

        v16h A[2];
        #pragma unroll
        for (int m = 0; m < 2; ++m)
            A[m] = ld_a16(Pbuf + (m * 16 + l16) * PSTR + kc * KC + half * 8);

        #pragma unroll
        for (int nt = 0; nt < 6; ++nt) {
            v16h B = ld_b16(Vrow + half * 8 * VSTR + (wave * 6 + nt) * 16 + l16, VSTR);
            accO[0][nt] = __builtin_amdgcn_wmma_f32_16x16x32_f16(
                false, A[0], false, B, (short)0, accO[0][nt], false, false);
            accO[1][nt] = __builtin_amdgcn_wmma_f32_16x16x32_f16(
                false, A[1], false, B, (short)0, accO[1][nt], false, false);
        }
    }

    // ---------------- write even rows (alphas == 1 exactly), zero odd rows ----------------
    float* ybase = y + segbase;
    #pragma unroll
    for (int m = 0; m < 2; ++m)
        #pragma unroll
        for (int nt = 0; nt < 6; ++nt) {
            int col = (wave * 6 + nt) * 16 + l16;
            #pragma unroll
            for (int v = 0; v < 8; ++v) {
                int qrow = q0 + m * 16 + half * 8 + v;
                ybase[(size_t)(2 * qrow) * DDIM + col] = accO[m][nt][v];
            }
        }
    {
        const int r  = t >> 3;
        const int cg = t & 7;
        float4 z = {0.f, 0.f, 0.f, 0.f};
        float4* dst = (float4*)(ybase + (size_t)(2 * (q0 + r) + 1) * DDIM);
        #pragma unroll
        for (int i = 0; i < 24; ++i) dst[cg + i * 8] = z;
    }
}

extern "C" void kernel_launch(void* const* d_in, const int* in_sizes, int n_in,
                              void* d_out, int out_size, void* d_ws, size_t ws_size,
                              hipStream_t stream) {
    (void)in_sizes; (void)n_in; (void)d_ws; (void)ws_size; (void)out_size;
    const float* x = (const float*)d_in[0];
    float* y = (float*)d_out;
    dilated_attn_f16_wmma<<<512, 256, 0, stream>>>(x, y);
}
